// DQN_5231270166668
// MI455X (gfx1250) — matrix operand order repaired and checked by reading, not compiled, not run on hardware
//
#include <hip/hip_runtime.h>
#include <hip/hip_bf16.h>
#include <math.h>

// ---------------------------------------------------------------------------
// DQN transformer forward, algebraically reduced:
// attention softmax is exactly uniform for ALL inputs (keys/values are
// row-constant by construction), so the output depends only on layer-3
// parameters:
//   sv  = sum_h val_p[3,0,h,:]
//   vr  = Wv3 @ sv  + bv3          (Wv3 = in_proj_w[3][900:1350])
//   c1  = out_w3 @ vr + out_b3
//   c2  = relu(lin_w3 @ c1 + lin_b3)
//   logit[d] = c2[d] * (1/sqrt(1+1e-5)) * bn_g[3,0] + bn_b[3,0]   (d < 436)
//   out[b,0,:] = softmax(logit)  for every b
// Matvecs: v_wmma_f32_16x16x32_f16 fully unrolled (15 WMMAs/wave), x-vector
// staged LDS-direct via global_load_async_to_lds_b128 (ASYNCcnt path).
// ---------------------------------------------------------------------------

typedef __attribute__((ext_vector_type(16))) _Float16 v16h;
typedef __attribute__((ext_vector_type(8)))  float    v8f;

#define EMB    450
#define NACT   436
#define BATCH  256
#define KPAD   480        // 15 chunks of 32
#define NCHUNK 15

// ---- kernel 1: sv[d] = sum over 18 heads of val_p[3,0,h,d] ----------------
__global__ void dqn_headsum(const float* __restrict__ val_p,
                            float* __restrict__ sv) {
  int d = blockIdx.x * blockDim.x + threadIdx.x;
  if (d < EMB) {
    float s = 0.0f;
    #pragma unroll
    for (int h = 0; h < 18; ++h)
      s += val_p[(size_t)(3 * 18 + h) * EMB + d];
    sv[d] = s;
  }
}

// ---- kernel 2: y = op(W @ x + b), W is 450x450 row-major, via WMMA --------
// One wave per 16-row tile. A = W tile (16x32 f16, ISA layout), B = x chunk
// broadcast into all 16 columns, C accumulates f32.
// Branchless: x is zero-padded to KPAD in LDS, so A values at k>=450
// contribute exactly 0 -- only their addresses need clamping in-bounds.
// NOTE: x always points into d_ws with >=512 floats of in-bounds window.
__global__ __launch_bounds__(32)
void dqn_matvec_wmma(const float* __restrict__ W,
                     const float* __restrict__ x,
                     const float* __restrict__ b,
                     float* __restrict__ y,
                     int do_relu) {
  __shared__ __align__(16) float    xf[512];   // raw f32 stage (async target)
  __shared__ __align__(32) _Float16 xh[KPAD];  // converted, zero-padded
  const int lane = threadIdx.x;                // wave32, single wave per block
  const int row0 = blockIdx.x * 16;

  // ---- async LDS-direct stage of x (GLOBAL_LOAD_ASYNC_TO_LDS_B128) ----
  {
    const unsigned lds_base = (unsigned)(size_t)(&xf[0]);  // LDS byte offset
    #pragma unroll
    for (int it = 0; it < 4; ++it) {
      const int i = (lane + 32 * it) * 4;                  // float index, 16B units
      const unsigned lds = lds_base + 4u * (unsigned)i;
      const unsigned long long ga = (unsigned long long)(size_t)(x + i);
      asm volatile("global_load_async_to_lds_b128 %0, %1, off"
                   :: "v"(lds), "v"(ga) : "memory");
    }
    asm volatile("s_wait_asynccnt 0x0" ::: "memory");
  }
  // convert to f16, zero-pad beyond EMB (single wave: no barrier needed)
  #pragma unroll
  for (int it = 0; it < NCHUNK; ++it) {
    const int i = lane + 32 * it;                          // 0..479
    xh[i] = (i < EMB) ? (_Float16)xf[i] : (_Float16)0.0f;
  }
  __syncthreads();

  const int m     = min(row0 + (lane & 15), EMB - 1);      // clamped row
  const int hi    = lane >> 4;                             // 0 | 1
  const int khalf = hi * 8;                                // A half K-offset
  const float* Wrow = W + (size_t)m * EMB;                 // 8B aligned (m*1800)
  __builtin_prefetch(Wrow, 0, 3);                          // global_prefetch_b8

  v8f c = {};
  #pragma unroll
  for (int kk = 0; kk < NCHUNK; ++kk) {
    const int kbase = kk * 32;

    // A-matrix (16-bit 16x32 ISA layout), lane M = lane&15:
    //   halves 0..7  <- K = kbase + khalf      + {0..7}
    //   halves 8..15 <- K = kbase + 16 + khalf + {0..7}
    v16h a;
    #pragma unroll
    for (int g = 0; g < 2; ++g) {
      const int k0 = kbase + 16 * g + khalf;
      #pragma unroll
      for (int jj = 0; jj < 4; ++jj) {
        const int kg = min(k0 + 2 * jj, EMB - 2);  // clamp addr; value*0 if k>=450
        const float2 w2 = *(const float2*)(Wrow + kg);
        a[8 * g + 2 * jj]     = (_Float16)w2.x;
        a[8 * g + 2 * jj + 1] = (_Float16)w2.y;
      }
    }

    // B-matrix (32x16): halves 0..7 <- K=kbase+khalf+{0..7}, halves 8..15 <-
    // K=kbase+16+khalf+{0..7}; columns broadcast -> two 16B LDS reads.
    typedef __attribute__((ext_vector_type(8))) _Float16 v8h_t;
    const v8h_t b0 = *(const v8h_t*)(xh + kbase + khalf);
    const v8h_t b1 = *(const v8h_t*)(xh + kbase + 16 + khalf);
    const v16h bm = __builtin_shufflevector(b0, b1, 0,1,2,3,4,5,6,7,8,9,10,11,12,13,14,15);

    c = __builtin_amdgcn_wmma_f32_16x16x32_f16(
            /*neg_a=*/false, a, /*neg_b=*/false, bm,
            /*c_mod=*/(short)0, c, /*reuse_a=*/false, /*reuse_b=*/false);
  }

  // C layout: VGPR r, lanes 0-15 -> M=r, lanes 16-31 -> M=8+r (all N equal).
  if ((lane & 15) == 0) {
    const int base = row0 + hi * 8;
    #pragma unroll
    for (int r = 0; r < 8; ++r) {
      const int row = base + r;
      if (row < EMB) {
        float v = c[r] + b[row];
        if (do_relu) v = fmaxf(v, 0.0f);
        y[row] = v;
      }
    }
  }
}

// ---- kernel 3: batchnorm-scale + softmax over 436 logits ------------------
__global__ __launch_bounds__(512)
void dqn_softmax(const float* __restrict__ c2,
                 const float* __restrict__ bn_g,
                 const float* __restrict__ bn_b,
                 float* __restrict__ p) {
  __shared__ float red[512];
  const int t = threadIdx.x;
  const float inv_bn = 0.99999499998749994f;     // 1/sqrt(1+1e-5)
  const float g  = bn_g[3 * 63 + 0];
  const float bb = bn_b[3 * 63 + 0];

  float v = (t < NACT) ? (c2[t] * inv_bn * g + bb) : -INFINITY;
  red[t] = v;
  __syncthreads();
  for (int s = 256; s > 0; s >>= 1) {
    if (t < s) red[t] = fmaxf(red[t], red[t + s]);
    __syncthreads();
  }
  const float mx = red[0];
  __syncthreads();
  float e = (t < NACT) ? expf(v - mx) : 0.0f;
  red[t] = e;
  __syncthreads();
  for (int s = 256; s > 0; s >>= 1) {
    if (t < s) red[t] += red[t + s];
    __syncthreads();
  }
  const float inv = 1.0f / red[0];
  if (t < NACT) p[t] = e * inv;
}

// ---- kernel 4: broadcast probs to all 256 batch rows ----------------------
__global__ void dqn_bcast(const float* __restrict__ p,
                          float* __restrict__ out) {
  int i = blockIdx.x * blockDim.x + threadIdx.x;
  if (i < BATCH * NACT) out[i] = p[i % NACT];
}

// ---------------------------------------------------------------------------
extern "C" void kernel_launch(void* const* d_in, const int* in_sizes, int n_in,
                              void* d_out, int out_size, void* d_ws, size_t ws_size,
                              hipStream_t stream) {
  // setup_inputs() order:
  // 0:x 1:card_table 2:pe 3:in_proj_w 4:in_proj_b 5:out_w 6:out_b
  // 7:lin_w 8:lin_b 9:bn_g 10:bn_b 11:key_p 12:val_p
  const float* in_proj_w = (const float*)d_in[3];   // (4, 1350, 450)
  const float* in_proj_b = (const float*)d_in[4];   // (4, 1350)
  const float* out_w     = (const float*)d_in[5];   // (4, 450, 450)
  const float* out_b     = (const float*)d_in[6];   // (4, 450)
  const float* lin_w     = (const float*)d_in[7];   // (4, 450, 450)
  const float* lin_b     = (const float*)d_in[8];   // (4, 450)
  const float* bn_g      = (const float*)d_in[9];   // (4, 63)
  const float* bn_b      = (const float*)d_in[10];  // (4, 63)
  const float* val_p     = (const float*)d_in[12];  // (4, 1, 18, 450)

  // layer-3 parameter slices
  const float* Wv3   = in_proj_w + (size_t)3 * 1350 * EMB + (size_t)900 * EMB;
  const float* bv3   = in_proj_b + (size_t)3 * 1350 + 900;
  const float* Wout3 = out_w + (size_t)3 * EMB * EMB;
  const float* bout3 = out_b + (size_t)3 * EMB;
  const float* Wlin3 = lin_w + (size_t)3 * EMB * EMB;
  const float* blin3 = lin_b + (size_t)3 * EMB;

  // workspace layout (floats) -- each x-vector has a 512-float in-bounds
  // window after it (async stage reads 512 floats)
  float* ws = (float*)d_ws;
  float* sv = ws + 0;      // 450
  float* vr = ws + 512;    // 450
  float* c1 = ws + 1024;   // 450
  float* c2 = ws + 1536;   // 450
  float* p  = ws + 2048;   // 436

  const int tiles = (EMB + 15) / 16;  // 29 row tiles

  dqn_headsum<<<(EMB + 255) / 256, 256, 0, stream>>>(val_p, sv);
  dqn_matvec_wmma<<<tiles, 32, 0, stream>>>(Wv3,   sv, bv3,   vr, 0);
  dqn_matvec_wmma<<<tiles, 32, 0, stream>>>(Wout3, vr, bout3, c1, 0);
  dqn_matvec_wmma<<<tiles, 32, 0, stream>>>(Wlin3, c1, blin3, c2, 1);
  dqn_softmax<<<1, 512, 0, stream>>>(c2, bn_g, bn_b, p);
  dqn_bcast<<<(BATCH * NACT + 255) / 256, 256, 0, stream>>>(p, (float*)d_out);
}
